// ESBN_45389214384222
// MI455X (gfx1250) — matrix operand order repaired and checked by reading, not compiled, not run on hardware
//
#include <hip/hip_runtime.h>
#include <hip/hip_fp16.h>

// ---------------------------------------------------------------------------
// ESBN forward for MI455X (gfx1250, wave32, WMMA).
//  - convs/linears: implicit-GEMM via v_wmma_f32_16x16x32_f16
//  - A fragments: two ds_load_b128 per wave (k order used here, same as B, is
//    per-lane contiguous: k = 16*hi + i)
//  - B fragments: pre-packed in fragment order in global memory; two
//    coalesced 32B loads per wave per 16-col group (weights are L2-resident)
//  - LDS A tile double-buffered: next chunk's global load overlaps the WMMAs
// ---------------------------------------------------------------------------

typedef __attribute__((ext_vector_type(16))) _Float16 v16h;
typedef __attribute__((ext_vector_type(8)))  _Float16 v8h;
typedef __attribute__((ext_vector_type(8)))  float    v8f;

constexpr int T_  = 64;
constexpr int B_  = 256;
constexpr int TB_ = T_ * B_;          // 16384 samples
constexpr int VD_ = 64;
constexpr int KD_ = 64;
constexpr int HD_ = 512;
constexpr int OD_ = 4;
constexpr int G4_ = 4 * HD_;          // 2048 gate width
constexpr int KXP_ = 608;             // pad(65 + 512) to multiple of 32
constexpr int LDA_ = 48;              // LDS row stride (halves); 96B = 16B-aligned

__device__ __forceinline__ float sigm(float x) { return 1.0f / (1.0f + expf(-x)); }

__device__ __forceinline__ v8f wmma16x16x32(v16h a, v16h b, v8f c) {
  return __builtin_amdgcn_wmma_f32_16x16x32_f16(false, a, false, b, (short)0, c,
                                                false, false);
}

// A fragment 16x32 from LDS tile (row stride LDA_ halves, rows 16B-aligned):
// two ds_load_b128 per lane + one shuffle.
__device__ __forceinline__ v16h load_frag_a(const _Float16* As, int wm) {
  const int lane = threadIdx.x & 31;
  const int m = lane & 15;
  const int hi = lane >> 4;
  const v8h* p = (const v8h*)(As + (wm * 16 + m) * LDA_ + hi * 16);
  const v8h lo = p[0];   // k = 16*hi + 0..7
  const v8h h2 = p[1];   // k = 16*hi + 8..15
  return __builtin_shufflevector(lo, h2, 0, 1, 2, 3, 4, 5, 6, 7, 8, 9, 10, 11,
                                 12, 13, 14, 15);
}

// ---------------------------------------------------------------------------
// Implicit-GEMM convolution (k=4, stride=2, VALID), f16 out (+bias, opt ReLU).
// Block: 256 thr = 8 waves, tile 64(M) x 64(N), BK=32; wave -> 16x32 of C.
// wTf: fragment-packed weights [n_blk][k_blk][lane][i], halves.
// ---------------------------------------------------------------------------
template <int IC, int IH, int IW, int OC, int OH, int OW, bool IN_F32, bool RELU>
__global__ void conv_wmma_kernel(const void* __restrict__ in_v,
                                 const _Float16* __restrict__ wTf,
                                 const float* __restrict__ bias,
                                 _Float16* __restrict__ out) {  // (S, OC, OH, OW)
  constexpr int OP = OH * OW;
  constexpr int Kreal = IC * 16;
  constexpr int KP = (Kreal + 31) & ~31;
  constexpr int KB = KP / 32;
  static_assert(((long long)TB_ * OP) % 64 == 0, "M must be multiple of 64");
  __shared__ __align__(16) _Float16 As[2][64 * LDA_];

  const float*    inF = (const float*)in_v;
  const _Float16* inH = (const _Float16*)in_v;

  const int row0 = blockIdx.x * 64;
  const int tid  = threadIdx.x;
  const int lane = tid & 31;
  const int wave = tid >> 5;
  const int wm = wave >> 1;   // 0..3
  const int wn = wave & 1;    // 0..1

  // Per-thread A-staging coordinates (8 contiguous k per thread).
  const int rl = tid >> 2;        // 0..63  tile row
  const int cc = tid & 3;         // 0..3   8-half group
  const int r  = row0 + rl;
  const int s   = r / OP;
  const int pos = r % OP;
  const int oy  = pos / OW;
  const int ox  = pos % OW;
  const int ibase = s * (IC * IH * IW) + (oy * 2) * IW + ox * 2;

  auto gather = [&](int kb) -> v8h {
    v8h frag;
#pragma unroll
    for (int j = 0; j < 8; ++j) {
      const int k = kb * 32 + cc * 8 + j;
      _Float16 v = (_Float16)0.0f;
      if (KP == Kreal || k < Kreal) {
        const int ic  = k >> 4;
        const int rem = k & 15;
        const int kh  = rem >> 2;
        const int kw  = rem & 3;
        const int off = ibase + ic * (IH * IW) + kh * IW + kw;
        v = IN_F32 ? (_Float16)inF[off] : inH[off];
      }
      frag[j] = v;
    }
    return frag;
  };

  v8f acc0 = {};
  v8f acc1 = {};

  // prologue: stage chunk 0
  *(v8h*)&As[0][rl * LDA_ + cc * 8] = gather(0);

  for (int kb = 0; kb < KB; ++kb) {
    __syncthreads();  // As[kb&1] visible to all waves
    const bool more = (kb + 1 < KB);
    v8h av;
    if (more) av = gather(kb + 1);  // overlaps the WMMAs below

    const v16h a = load_frag_a(As[kb & 1], wm);
    const _Float16* bp =
        wTf + ((size_t)((2 * wn) * KB + kb) * 32 + lane) * 16;
    const v16h b0 = *(const v16h*)bp;                       // n_blk = 2*wn
    const v16h b1 = *(const v16h*)(bp + (size_t)KB * 512);  // n_blk = 2*wn+1
    __builtin_prefetch(bp + 512, 0, 1);  // speculative; past-end is dropped
    acc0 = wmma16x16x32(a, b0, acc0);
    acc1 = wmma16x16x32(a, b1, acc1);

    if (more) *(v8h*)&As[(kb + 1) & 1][rl * LDA_ + cc * 8] = av;
  }

  // ---- epilogue: bias (+ReLU), scatter to NCHW f16 (rows always in range)
  const int col0  = wn * 32 + (lane & 15);
  const int rbase = row0 + wm * 16 + ((lane >> 4) << 3);
#pragma unroll
  for (int e = 0; e < 8; ++e) {
    const int r2   = rbase + e;
    const int s2   = r2 / OP;
    const int pos2 = r2 % OP;
    if (col0 < OC) {
      float v = acc0[e] + bias[col0];
      if (RELU) v = fmaxf(v, 0.0f);
      out[(s2 * OC + col0) * OP + pos2] = (_Float16)v;
    }
    const int col1 = col0 + 16;
    if (col1 < OC) {
      float v = acc1[e] + bias[col1];
      if (RELU) v = fmaxf(v, 0.0f);
      out[(s2 * OC + col1) * OP + pos2] = (_Float16)v;
    }
  }
}

// ---------------------------------------------------------------------------
// Plain GEMM: C[M,N](f32) = A[M,K](f16 row-major, 16B-aligned rows) x
//             Bf (fragment-packed) + bias[N].
// M % 64 == 0, K % 32 == 0, N % 64 == 0, (lda*2) % 16 == 0.
// grid = (M/64, N/64), block = 256.
// ---------------------------------------------------------------------------
__global__ void gemm_f16f32_wmma(const _Float16* __restrict__ A, int lda, int K,
                                 const _Float16* __restrict__ Bf, int N,
                                 const float* __restrict__ bias,
                                 float* __restrict__ C) {
  __shared__ __align__(16) _Float16 As[2][64 * LDA_];
  const int KB = K / 32;

  const int row0 = blockIdx.x * 64;
  const int n0   = blockIdx.y * 64;
  const int nblk0 = blockIdx.y * 4;
  const int tid  = threadIdx.x;
  const int lane = tid & 31;
  const int wave = tid >> 5;
  const int wm = wave >> 1;
  const int wn = wave & 1;

  const int rl = tid >> 2;
  const int cc = tid & 3;
  const _Float16* arow = A + (size_t)(row0 + rl) * lda + cc * 8;

  v8f acc0 = {};
  v8f acc1 = {};

  // prologue: stage chunk 0 (one global_load_b128 + one ds_store_b128)
  *(v8h*)&As[0][rl * LDA_ + cc * 8] = *(const v8h*)arow;

  for (int kb = 0; kb < KB; ++kb) {
    __syncthreads();
    const bool more = (kb + 1 < KB);
    v8h av;
    if (more) av = *(const v8h*)(arow + (kb + 1) * 32);  // overlaps WMMAs

    const v16h a = load_frag_a(As[kb & 1], wm);
    const _Float16* bp =
        Bf + ((size_t)((nblk0 + 2 * wn) * KB + kb) * 32 + lane) * 16;
    const v16h b0 = *(const v16h*)bp;
    const v16h b1 = *(const v16h*)(bp + (size_t)KB * 512);
    __builtin_prefetch(bp + 512, 0, 1);
    acc0 = wmma16x16x32(a, b0, acc0);
    acc1 = wmma16x16x32(a, b1, acc1);

    if (more) *(v8h*)&As[(kb + 1) & 1][rl * LDA_ + cc * 8] = av;
  }

  const int col0  = n0 + wn * 32 + (lane & 15);
  const int rbase = row0 + wm * 16 + ((lane >> 4) << 3);
#pragma unroll
  for (int e = 0; e < 8; ++e) {
    const int r2 = rbase + e;
    C[(size_t)r2 * N + col0]      = acc0[e] + bias[col0];
    C[(size_t)r2 * N + col0 + 16] = acc1[e] + bias[col0 + 16];
  }
}

// ---------------------------------------------------------------------------
// Weight packing into fragment order: out[((n_blk*KB + k_blk)*32 + lane)*16+i]
// holds W[n = n_blk*16 + (lane&15)][k = k_blk*32 + 16*(lane>>4) + i].
// ---------------------------------------------------------------------------
__global__ void pack_conv_w_frag(const float* __restrict__ w, int OC, int IC,
                                 int KP, _Float16* __restrict__ out) {
  const int KB = KP / 32;
  const int total = 4 * KB * 512;  // 4 n_blks of 16 cols
  const int idx = blockIdx.x * blockDim.x + threadIdx.x;
  if (idx >= total) return;
  const int i    = idx & 15;
  const int lane = (idx >> 4) & 31;
  const int t2   = idx >> 9;
  const int kb   = t2 % KB;
  const int nb   = t2 / KB;
  const int n = nb * 16 + (lane & 15);
  const int k = kb * 32 + (lane >> 4) * 16 + i;
  float v = 0.0f;
  if (k < IC * 16 && n < OC) {
    const int ic = k >> 4;
    const int rem = k & 15;
    v = w[(n * IC + ic) * 16 + rem];  // (OC, IC, 4, 4)
  }
  out[idx] = (_Float16)v;
}

__global__ void pack_lin_frag(const float* __restrict__ W, int Nout, int K,
                              int NB, _Float16* __restrict__ out) {
  const int KB = K / 32;
  const int total = NB * KB * 512;
  const int idx = blockIdx.x * blockDim.x + threadIdx.x;
  if (idx >= total) return;
  const int i    = idx & 15;
  const int lane = (idx >> 4) & 31;
  const int t2   = idx >> 9;
  const int kb   = t2 % KB;
  const int nb   = t2 / KB;
  const int n = nb * 16 + (lane & 15);
  const int k = kb * 32 + (lane >> 4) * 16 + i;
  float v = 0.0f;
  if (n < Nout && k < K) v = W[n * K + k];  // W is (Nout, K) row-major
  out[idx] = (_Float16)v;
}

__global__ void pack_wcat_frag(const float* __restrict__ w_ih,  // (2048, 65)
                               const float* __restrict__ w_hh,  // (2048, 512)
                               _Float16* __restrict__ out) {    // KXP_ x 2048
  const int KB = KXP_ / 32;            // 19
  const int total = (G4_ / 16) * KB * 512;
  const int idx = blockIdx.x * blockDim.x + threadIdx.x;
  if (idx >= total) return;
  const int i    = idx & 15;
  const int lane = (idx >> 4) & 31;
  const int t2   = idx >> 9;
  const int kb   = t2 % KB;
  const int nb   = t2 / KB;
  const int n = nb * 16 + (lane & 15);
  const int k = kb * 32 + (lane >> 4) * 16 + i;
  float v = 0.0f;
  if (k < 65) v = w_ih[n * 65 + k];
  else if (k < 577) v = w_hh[n * 512 + (k - 65)];
  out[idx] = (_Float16)v;
}

__global__ void combine_bias(const float* __restrict__ b_ih,
                             const float* __restrict__ b_hh,
                             float* __restrict__ bc) {
  const int j = blockIdx.x * blockDim.x + threadIdx.x;
  if (j < G4_) bc[j] = b_ih[j] + b_hh[j];
}

// ---------------------------------------------------------------------------
// LSTM pointwise; also mirrors hx into xh (f16, gate GEMM input) and h16
// (dedicated 16B-aligned f16 buffer for the key-head GEMM).
// ---------------------------------------------------------------------------
__global__ void lstm_pointwise(const float* __restrict__ gates,
                               float* __restrict__ hx, float* __restrict__ cx,
                               _Float16* __restrict__ xh,
                               _Float16* __restrict__ h16) {
  const int idx = blockIdx.x * blockDim.x + threadIdx.x;  // B_*HD_
  if (idx >= B_ * HD_) return;
  const int b = idx >> 9;
  const int j = idx & 511;
  const float* gr = gates + (size_t)b * G4_;
  const float i = sigm(gr[j]);
  const float f = sigm(gr[512 + j]);
  const float g = tanhf(gr[1024 + j]);
  const float o = sigm(gr[1536 + j]);
  const float c2 = f * cx[idx] + i * g;
  const float h = o * tanhf(c2);
  cx[idx] = c2;
  hx[idx] = h;
  xh[(size_t)b * KXP_ + 65 + j] = (_Float16)h;
  h16[idx] = (_Float16)h;
}

// ---------------------------------------------------------------------------
// Output + gate heads: y[t,b,:4] and g[b]; 512-long dots (tiny, VALU).
// ---------------------------------------------------------------------------
__global__ void heads_kernel(int t, const float* __restrict__ hx,
                             const float* __restrict__ out_w,
                             const float* __restrict__ out_b,
                             const float* __restrict__ gate_w,
                             const float* __restrict__ gate_b,
                             float* __restrict__ y_out,
                             float* __restrict__ gvec) {
  const int idx = blockIdx.x * blockDim.x + threadIdx.x;  // B_*5
  if (idx >= B_ * 5) return;
  const int b = idx / 5;
  const int o = idx % 5;
  const float* h = hx + (size_t)b * HD_;
  const float* w = (o < 4) ? out_w + o * HD_ : gate_w;
  float s = 0.0f;
  for (int j = 0; j < HD_; ++j) s += h[j] * w[j];
  if (o < 4) y_out[((size_t)t * B_ + b) * OD_ + o] = s + out_b[o];
  else       gvec[b] = s + gate_b[0];
}

// ---------------------------------------------------------------------------
// Content-addressed memory read + append. One block per batch element.
// t==0 path yields zero kx (matches reference's forced zero).
// ---------------------------------------------------------------------------
__global__ void attention_kernel(int t, const float* __restrict__ z_all,
                                 const float* __restrict__ kw,
                                 const float* __restrict__ gvec,
                                 const float* __restrict__ conf_w,
                                 const float* __restrict__ conf_b,
                                 float* __restrict__ Mk, float* __restrict__ Mv,
                                 _Float16* __restrict__ xh) {
  const int b = blockIdx.x;
  const int tid = threadIdx.x;  // 128
  __shared__ float sim[T_], wk[T_], ck[T_], invs;
  const float* z = z_all + ((size_t)t * B_ + b) * VD_;

  if (tid < T_) {
    const float* mv = Mv + ((size_t)b * T_ + tid) * VD_;
    float s = 0.0f;
    for (int d = 0; d < VD_; ++d) s += mv[d] * z[d];
    sim[tid] = s;
    ck[tid] = sigm(s * conf_w[0] + conf_b[0]);
  }
  __syncthreads();
  if (tid == 0) {
    float mx = -1e30f;
    for (int n = 0; n < t; ++n) mx = fmaxf(mx, sim[n]);
    float sum = 0.0f;
    for (int n = 0; n < T_; ++n) {
      const float e = (n < t) ? expf(sim[n] - mx) : 0.0f;
      wk[n] = e;
      sum += e;
    }
    invs = (sum > 0.0f) ? (1.0f / sum) : 0.0f;
  }
  __syncthreads();
  const float inv = invs;
  if (tid < KD_ + 1) {
    float acc = 0.0f;
    for (int n = 0; n < T_; ++n) {
      const float memv =
          (tid < KD_) ? Mk[((size_t)b * T_ + n) * KD_ + tid] : ck[n];
      acc += wk[n] * inv * memv;
    }
    const float kx = sigm(gvec[b]) * acc;
    xh[(size_t)b * KXP_ + tid] = (_Float16)kx;
  }
  __syncthreads();  // appends below must not race the Mk reads above
  if (tid < KD_) Mk[((size_t)b * T_ + t) * KD_ + tid] = kw[(size_t)b * KD_ + tid];
  if (tid < VD_) Mv[((size_t)b * T_ + t) * VD_ + tid] = z[tid];
}

// ---------------------------------------------------------------------------
// Host-side orchestration.
// ---------------------------------------------------------------------------
extern "C" void kernel_launch(void* const* d_in, const int* in_sizes, int n_in,
                              void* d_out, int out_size, void* d_ws,
                              size_t ws_size, hipStream_t stream) {
  (void)in_sizes; (void)n_in; (void)out_size; (void)ws_size;

  const float* images  = (const float*)d_in[0];
  const float* conv1_w = (const float*)d_in[1];
  const float* conv1_b = (const float*)d_in[2];
  const float* conv2_w = (const float*)d_in[3];
  const float* conv2_b = (const float*)d_in[4];
  const float* conv3_w = (const float*)d_in[5];
  const float* conv3_b = (const float*)d_in[6];
  const float* enc_w   = (const float*)d_in[7];
  const float* enc_b   = (const float*)d_in[8];
  const float* w_ih    = (const float*)d_in[9];
  const float* w_hh    = (const float*)d_in[10];
  const float* b_ih    = (const float*)d_in[11];
  const float* b_hh    = (const float*)d_in[12];
  const float* out_w   = (const float*)d_in[13];
  const float* out_b   = (const float*)d_in[14];
  const float* gate_w  = (const float*)d_in[15];
  const float* gate_b  = (const float*)d_in[16];
  const float* key_w   = (const float*)d_in[17];
  const float* key_b   = (const float*)d_in[18];
  const float* conf_w  = (const float*)d_in[19];
  const float* conf_b  = (const float*)d_in[20];
  float* y_out = (float*)d_out;

  // ---- workspace carve-up (256B aligned slices) ----
  char* base = (char*)d_ws;
  size_t off = 0;
  auto alloc = [&](size_t bytes) -> void* {
    void* p = base + off;
    off += (bytes + 255) & ~(size_t)255;
    return p;
  };
  _Float16* wT1   = (_Float16*)alloc((size_t)64 * 64 * 2);     // 4 nb * 2 kb
  _Float16* wT2   = (_Float16*)alloc((size_t)512 * 64 * 2);
  _Float16* wT3   = (_Float16*)alloc((size_t)1024 * 64 * 2);
  _Float16* wTenc = (_Float16*)alloc((size_t)256 * 64 * 2);
  _Float16* wTcat = (_Float16*)alloc((size_t)KXP_ * G4_ * 2);
  _Float16* wTkey = (_Float16*)alloc((size_t)512 * 64 * 2);
  float*    bcat  = (float*)alloc((size_t)G4_ * 4);
  _Float16* act1  = (_Float16*)alloc((size_t)TB_ * 32 * 225 * 2);
  _Float16* act2  = (_Float16*)alloc((size_t)TB_ * 64 * 36 * 2);
  _Float16* act3  = (_Float16*)alloc((size_t)TB_ * 256 * 2);
  float*    z_all = (float*)alloc((size_t)TB_ * VD_ * 4);
  _Float16* xh    = (_Float16*)alloc((size_t)B_ * KXP_ * 2);
  _Float16* h16   = (_Float16*)alloc((size_t)B_ * HD_ * 2);
  float*    gates = (float*)alloc((size_t)B_ * G4_ * 4);
  float*    hx    = (float*)alloc((size_t)B_ * HD_ * 4);
  float*    cx    = (float*)alloc((size_t)B_ * HD_ * 4);
  float*    kwbuf = (float*)alloc((size_t)B_ * KD_ * 4);
  float*    gvec  = (float*)alloc((size_t)B_ * 4);
  float*    Mk    = (float*)alloc((size_t)B_ * T_ * KD_ * 4);
  float*    Mv    = (float*)alloc((size_t)B_ * T_ * VD_ * 4);

  // ---- zero recurrent state (graph-capture-safe async memsets) ----
  hipMemsetAsync(xh, 0, (size_t)B_ * KXP_ * 2, stream);
  hipMemsetAsync(hx, 0, (size_t)B_ * HD_ * 4, stream);
  hipMemsetAsync(cx, 0, (size_t)B_ * HD_ * 4, stream);
  hipMemsetAsync(Mk, 0, (size_t)B_ * T_ * KD_ * 4, stream);
  hipMemsetAsync(Mv, 0, (size_t)B_ * T_ * VD_ * 4, stream);

  // ---- pack weights into fragment-ready f16 operands ----
  pack_conv_w_frag<<<(64 * 64 + 255) / 256, 256, 0, stream>>>(conv1_w, 32, 3,
                                                              64, wT1);
  pack_conv_w_frag<<<(512 * 64 + 255) / 256, 256, 0, stream>>>(conv2_w, 64, 32,
                                                               512, wT2);
  pack_conv_w_frag<<<(1024 * 64 + 255) / 256, 256, 0, stream>>>(conv3_w, 64, 64,
                                                                1024, wT3);
  pack_lin_frag<<<(256 * 64 + 255) / 256, 256, 0, stream>>>(enc_w, 64, 256, 4,
                                                            wTenc);
  pack_lin_frag<<<(512 * 64 + 255) / 256, 256, 0, stream>>>(key_w, 64, 512, 4,
                                                            wTkey);
  pack_wcat_frag<<<(KXP_ * G4_ + 255) / 256, 256, 0, stream>>>(w_ih, w_hh,
                                                               wTcat);
  combine_bias<<<(G4_ + 255) / 256, 256, 0, stream>>>(b_ih, b_hh, bcat);

  // ---- encoder (all T*B samples at once) ----
  {
    const int m1 = TB_ * 225;
    conv_wmma_kernel<3, 32, 32, 32, 15, 15, true, true>
        <<<m1 / 64, 256, 0, stream>>>(images, wT1, conv1_b, act1);
    const int m2 = TB_ * 36;
    conv_wmma_kernel<32, 15, 15, 64, 6, 6, false, true>
        <<<m2 / 64, 256, 0, stream>>>(act1, wT2, conv2_b, act2);
    const int m3 = TB_ * 4;
    conv_wmma_kernel<64, 6, 6, 64, 2, 2, false, false>
        <<<m3 / 64, 256, 0, stream>>>(act2, wT3, conv3_b, act3);
    // enc linear: (TB,256) x (256,64) -> z_all (T,B,64) f32
    dim3 ge(TB_ / 64, 1);
    gemm_f16f32_wmma<<<ge, 256, 0, stream>>>(act3, 256, 256, wTenc, 64, enc_b,
                                             z_all);
  }

  // ---- recurrent loop: 64 sequential steps ----
  for (int t = 0; t < T_; ++t) {
    dim3 gg(B_ / 64, G4_ / 64);
    gemm_f16f32_wmma<<<gg, 256, 0, stream>>>(xh, KXP_, KXP_, wTcat, G4_, bcat,
                                             gates);
    lstm_pointwise<<<(B_ * HD_ + 255) / 256, 256, 0, stream>>>(gates, hx, cx,
                                                               xh, h16);
    dim3 gk(B_ / 64, KD_ / 64);
    gemm_f16f32_wmma<<<gk, 256, 0, stream>>>(h16, HD_, HD_, wTkey, KD_, key_b,
                                             kwbuf);
    heads_kernel<<<(B_ * 5 + 255) / 256, 256, 0, stream>>>(
        t, hx, out_w, out_b, gate_w, gate_b, y_out, gvec);
    attention_kernel<<<B_, 128, 0, stream>>>(t, z_all, kwbuf, gvec, conf_w,
                                             conf_b, Mk, Mv, xh);
  }
}
